// MultiChannelAttention_59760174956707
// MI455X (gfx1250) — matrix operand order repaired and checked by reading, not compiled, not run on hardware
//
#include <hip/hip_runtime.h>
#include <hip/hip_bf16.h>
#include <stdint.h>

// ---------------------------------------------------------------------------
// MI455X (gfx1250) attention layer, bf16 WMMA pipeline:
//   conv f32->bf16 -> [Q,K,V proj + RoPE] -> S=QK^T (causal, f32)
//   -> row softmax -> O = P@V -> out = O@Wf^T + bf (f32)
// All matmuls use v_wmma_f32_16x16x32_bf16 through one 128x128x32 tile kernel.
// ---------------------------------------------------------------------------

typedef unsigned short ushort_t;
typedef __attribute__((ext_vector_type(16))) __bf16 v16bf;
typedef __attribute__((ext_vector_type(8)))  float  v8f;

union FragBF { uint4 q[2]; v16bf v; };
union Acc8   { v8f v; float f[8]; };

__device__ __forceinline__ ushort_t f2bf(float x) {
  union { float f; unsigned int u; } c; c.f = x;
  unsigned int r = c.u + 0x7FFFu + ((c.u >> 16) & 1u);   // RNE
  return (ushort_t)(r >> 16);
}

// ---------------------------------------------------------------------------
// f32 -> bf16 bulk convert (8 elems / thread, 32B read, 16B write)
// ---------------------------------------------------------------------------
__global__ __launch_bounds__(256) void conv_f32_to_bf16(
    const float* __restrict__ in, ushort_t* __restrict__ out, long long n8) {
  long long i = (long long)blockIdx.x * 256 + threadIdx.x;
  if (i >= n8) return;
  const float4* p = (const float4*)in + 2 * i;
  float4 a = p[0], b = p[1];
  uint4 o;
  o.x = f2bf(a.x) | ((unsigned)f2bf(a.y) << 16);
  o.y = f2bf(a.z) | ((unsigned)f2bf(a.w) << 16);
  o.z = f2bf(b.x) | ((unsigned)f2bf(b.y) << 16);
  o.w = f2bf(b.z) | ((unsigned)f2bf(b.w) << 16);
  ((uint4*)out)[i] = o;
}

// ---------------------------------------------------------------------------
// Tiled bf16 GEMM: C(M x N) = A(M x K) @ B(N x K)^T  (both K-major)
// Block 256 threads = 8 waves; tile 128x128, K-step 32.
// Wave (wr in 0..3, wc in 0..1) owns 32x64 -> 2x4 grid of 16x16 WMMA tiles.
// Epilogue modes:
//  0: +bias, RoPE(t = row & 2047), store bf16 row-major       (Q, K)
//  1: +bias, store bf16 TRANSPOSED per batch: Vt[b][n][t]     (V)
//  2: *scale, causal mask (col>row -> -1e30), store f32       (scores)
//  3: store bf16 row-major (batch-strided)                    (O = P@V)
//  4: +bias, store f32 row-major                              (final out)
// ---------------------------------------------------------------------------
__global__ __launch_bounds__(256) void gemm_bf16_128x128(
    const ushort_t* __restrict__ A, const ushort_t* __restrict__ B,
    const float* __restrict__ bias,
    ushort_t* __restrict__ outB, float* __restrict__ outF,
    int N, int K,
    long long aBatch, long long bBatch, long long oBatch,
    float scale, int mode)
{
  const int tid    = threadIdx.x;
  const int m_base = blockIdx.y * 128;
  const int n_base = blockIdx.x * 128;
  const int z      = blockIdx.z;
  A += (size_t)z * aBatch;
  B += (size_t)z * bBatch;

  if (mode == 2 && n_base > m_base + 127) {
    // tile entirely above the causal diagonal: fill -1e30 and exit
    int r = tid >> 1, hc = (tid & 1) * 64;
    float4 neg = make_float4(-1e30f, -1e30f, -1e30f, -1e30f);
    float4* o4 = (float4*)(outF + (size_t)z * oBatch +
                           (size_t)(m_base + r) * N + n_base + hc);
#pragma unroll
    for (int j = 0; j < 16; ++j) o4[j] = neg;
    return;
  }

  // LDS tiles, row stride 40 bf16 (80B) -> conflict-free b128 reads
  __shared__ ushort_t As[128 * 40];
  __shared__ ushort_t Bs[128 * 40];

  const int row = tid >> 1;
  const int hf  = tid & 1;
  const uint4* ga = (const uint4*)(A + (size_t)(m_base + row) * K + hf * 16);
  const uint4* gb = (const uint4*)(B + (size_t)(n_base + row) * K + hf * 16);
  uint4* sa = (uint4*)(&As[row * 40 + hf * 16]);
  uint4* sb = (uint4*)(&Bs[row * 40 + hf * 16]);

  const int lane = tid & 31;
  const int wave = tid >> 5;
  const int wr = wave & 3;   // 4 slices of 32 rows
  const int wc = wave >> 1 >> 1; // == wave >> 2 : 2 slices of 64 cols
  const int lr = lane & 15;
  const int lh = lane >> 4;

  const v8f vzero = {0.f, 0.f, 0.f, 0.f, 0.f, 0.f, 0.f, 0.f};
  v8f acc[2][4];
#pragma unroll
  for (int mi = 0; mi < 2; ++mi)
#pragma unroll
    for (int ni = 0; ni < 4; ++ni) acc[mi][ni] = vzero;

  const char* AsB = (const char*)As;
  const char* BsB = (const char*)Bs;

  for (int k0 = 0; k0 < K; k0 += 32) {
    __syncthreads();
    sa[0] = ga[0]; sa[1] = ga[1];
    sb[0] = gb[0]; sb[1] = gb[1];
    ga += 4; gb += 4;
    __builtin_prefetch(ga, 0, 0);   // global_prefetch_b8 for next K tile
    __builtin_prefetch(gb, 0, 0);
    __syncthreads();

    FragBF af[2], bfr[4];
    // A frag (ISA 16-bit 16x32): lane holds row M=lr, k chunks [lh*8,+8) and [16+lh*8,+8)
#pragma unroll
    for (int mi = 0; mi < 2; ++mi) {
      const uint4* pa = (const uint4*)(AsB + (wr * 32 + mi * 16 + lr) * 80 + lh * 16);
      af[mi].q[0] = pa[0];
      af[mi].q[1] = pa[2];   // +32 bytes
    }
    // B frag (32x16): lane holds col N=lr, k chunks [lh*8,+8) and [16+lh*8,+8)
#pragma unroll
    for (int ni = 0; ni < 4; ++ni) {
      const uint4* pb = (const uint4*)(BsB + (wc * 64 + ni * 16 + lr) * 80 + lh * 16);
      bfr[ni].q[0] = pb[0];
      bfr[ni].q[1] = pb[2];
    }
#pragma unroll
    for (int mi = 0; mi < 2; ++mi)
#pragma unroll
      for (int ni = 0; ni < 4; ++ni)
        acc[mi][ni] = __builtin_amdgcn_wmma_f32_16x16x32_bf16(
            false, af[mi].v, false, bfr[ni].v, (short)0, acc[mi][ni],
            false, false);
  }

  // ---------------- epilogue ----------------
#pragma unroll
  for (int mi = 0; mi < 2; ++mi) {
#pragma unroll
    for (int ni = 0; ni < 4; ++ni) {
      Acc8 c; c.v = acc[mi][ni];
      const int gc = n_base + wc * 64 + ni * 16 + lr;         // D col = lane&15
      const float bv = bias ? bias[gc] : 0.f;
      const int m0 = m_base + wr * 32 + mi * 16 + lh * 8;     // D row = lh*8 + v

      if (mode == 0) {                  // RoPE + bf16 row-major (Q, K)
#pragma unroll
        for (int v = 0; v < 8; ++v) {
          int gr = m0 + v;
          int t  = gr & 2047;           // position within batch
          float val = c.f[v] + bv;
          float partner = __shfl_xor(val, 1, 32);    // interleaved pair = col^1
          float freq = __expf(-(float)(gc & ~1) * (9.2103403719761836f / 1024.f));
          float ang  = (float)t * freq;
          float cs = cosf(ang), sn = sinf(ang);
          val = (gc & 1) ? fmaf(partner, sn, val * cs)    // x2*cos + x1*sin
                         : fmaf(-partner, sn, val * cs);  // x1*cos - x2*sin
          outB[(size_t)gr * N + gc] = f2bf(val);
        }
      } else if (mode == 1) {           // V transposed: Vt[b][gc][t], 8 consecutive t
        int b  = m0 >> 11;
        int t0 = m0 & 2047;
        uint4 o;
        o.x = f2bf(c.f[0] + bv) | ((unsigned)f2bf(c.f[1] + bv) << 16);
        o.y = f2bf(c.f[2] + bv) | ((unsigned)f2bf(c.f[3] + bv) << 16);
        o.z = f2bf(c.f[4] + bv) | ((unsigned)f2bf(c.f[5] + bv) << 16);
        o.w = f2bf(c.f[6] + bv) | ((unsigned)f2bf(c.f[7] + bv) << 16);
        *(uint4*)(outB + ((size_t)b * 1024 + gc) * 2048 + t0) = o;
      } else if (mode == 2) {           // scaled causal scores, f32
#pragma unroll
        for (int v = 0; v < 8; ++v) {
          int gr = m0 + v;
          float val = c.f[v] * scale;
          if (gc > gr) val = -1e30f;
          outF[(size_t)z * oBatch + (size_t)gr * N + gc] = val;
        }
      } else if (mode == 3) {           // plain bf16 row-major (batch-strided)
#pragma unroll
        for (int v = 0; v < 8; ++v) {
          int gr = m0 + v;
          outB[(size_t)z * oBatch + (size_t)gr * N + gc] = f2bf(c.f[v]);
        }
      } else {                          // bias + f32 row-major (final output)
#pragma unroll
        for (int v = 0; v < 8; ++v) {
          int gr = m0 + v;
          outF[(size_t)gr * N + gc] = c.f[v] + bv;
        }
      }
    }
  }
}

// ---------------------------------------------------------------------------
// Row softmax over 2048 f32 -> bf16 probabilities. One block per row.
// ---------------------------------------------------------------------------
__device__ __forceinline__ float waveMax(float v) {
#pragma unroll
  for (int o = 16; o > 0; o >>= 1) v = fmaxf(v, __shfl_xor(v, o, 32));
  return v;
}
__device__ __forceinline__ float waveSum(float v) {
#pragma unroll
  for (int o = 16; o > 0; o >>= 1) v += __shfl_xor(v, o, 32);
  return v;
}

__global__ __launch_bounds__(256) void softmax_rows(
    const float* __restrict__ S, ushort_t* __restrict__ P) {
  __shared__ float red[8];
  const int tid = threadIdx.x;
  const size_t rowOff = (size_t)blockIdx.x * 2048;
  const float4* s4 = (const float4*)(S + rowOff);
  float4 x0 = s4[tid];
  float4 x1 = s4[tid + 256];

  float m = fmaxf(fmaxf(fmaxf(x0.x, x0.y), fmaxf(x0.z, x0.w)),
                  fmaxf(fmaxf(x1.x, x1.y), fmaxf(x1.z, x1.w)));
  m = waveMax(m);
  if ((tid & 31) == 0) red[tid >> 5] = m;
  __syncthreads();
  float rowMax = red[0];
#pragma unroll
  for (int i = 1; i < 8; ++i) rowMax = fmaxf(rowMax, red[i]);
  __syncthreads();

  float e0x = __expf(x0.x - rowMax), e0y = __expf(x0.y - rowMax);
  float e0z = __expf(x0.z - rowMax), e0w = __expf(x0.w - rowMax);
  float e1x = __expf(x1.x - rowMax), e1y = __expf(x1.y - rowMax);
  float e1z = __expf(x1.z - rowMax), e1w = __expf(x1.w - rowMax);
  float s = (e0x + e0y + e0z + e0w) + (e1x + e1y + e1z + e1w);
  s = waveSum(s);
  if ((tid & 31) == 0) red[tid >> 5] = s;
  __syncthreads();
  float rowSum = 0.f;
#pragma unroll
  for (int i = 0; i < 8; ++i) rowSum += red[i];
  float inv = 1.0f / rowSum;

  uint2* p2 = (uint2*)(P + rowOff);
  uint2 o0, o1;
  o0.x = f2bf(e0x * inv) | ((unsigned)f2bf(e0y * inv) << 16);
  o0.y = f2bf(e0z * inv) | ((unsigned)f2bf(e0w * inv) << 16);
  o1.x = f2bf(e1x * inv) | ((unsigned)f2bf(e1y * inv) << 16);
  o1.y = f2bf(e1z * inv) | ((unsigned)f2bf(e1w * inv) << 16);
  p2[tid]       = o0;
  p2[tid + 256] = o1;
}

// ---------------------------------------------------------------------------
extern "C" void kernel_launch(void* const* d_in, const int* in_sizes, int n_in,
                              void* d_out, int out_size, void* d_ws, size_t ws_size,
                              hipStream_t stream) {
  const float* x   = (const float*)d_in[0];
  const float* Wq  = (const float*)d_in[1];
  const float* bq  = (const float*)d_in[2];
  const float* Wk  = (const float*)d_in[3];
  const float* bk  = (const float*)d_in[4];
  const float* Wv  = (const float*)d_in[5];
  const float* bv  = (const float*)d_in[6];
  const float* Wf  = (const float*)d_in[7];
  const float* bfp = (const float*)d_in[8];

  const int Bz = 4, T = 2048, C = 1024;
  const int M = Bz * T;                      // 8192

  uint8_t* ws = (uint8_t*)d_ws;
  size_t off = 0;
  auto alloc = [&](size_t bytes) {
    void* p = ws + off;
    off += (bytes + 255) & ~(size_t)255;
    return p;
  };

  ushort_t* xbf  = (ushort_t*)alloc((size_t)M * C * 2);   // 16 MB
  ushort_t* wqbf = (ushort_t*)alloc((size_t)C * C * 2);
  ushort_t* wkbf = (ushort_t*)alloc((size_t)C * C * 2);
  ushort_t* wvbf = (ushort_t*)alloc((size_t)C * C * 2);
  ushort_t* wfbf = (ushort_t*)alloc((size_t)C * C * 2);
  ushort_t* Qbf  = (ushort_t*)alloc((size_t)M * C * 2);
  ushort_t* Kbf  = (ushort_t*)alloc((size_t)M * C * 2);
  ushort_t* Vt   = (ushort_t*)alloc((size_t)M * C * 2);   // [B][C][T]
  float*    Sf   = (float*)   alloc((size_t)Bz * T * T * 4); // 64 MB
  ushort_t* Pbf  = (ushort_t*)alloc((size_t)Bz * T * T * 2); // 32 MB
  ushort_t* Obf  = (ushort_t*)alloc((size_t)M * C * 2);
  (void)ws_size; (void)in_sizes; (void)n_in; (void)out_size;

  // 1) bf16 conversion
  conv_f32_to_bf16<<<(M * C / 8) / 256, 256, 0, stream>>>(x,  xbf,  (long long)M * C / 8);
  conv_f32_to_bf16<<<(C * C / 8) / 256, 256, 0, stream>>>(Wq, wqbf, (long long)C * C / 8);
  conv_f32_to_bf16<<<(C * C / 8) / 256, 256, 0, stream>>>(Wk, wkbf, (long long)C * C / 8);
  conv_f32_to_bf16<<<(C * C / 8) / 256, 256, 0, stream>>>(Wv, wvbf, (long long)C * C / 8);
  conv_f32_to_bf16<<<(C * C / 8) / 256, 256, 0, stream>>>(Wf, wfbf, (long long)C * C / 8);

  // 2) projections (+RoPE for Q/K, transposed store for V)
  dim3 gProj(C / 128, M / 128, 1);
  gemm_bf16_128x128<<<gProj, 256, 0, stream>>>(xbf, wqbf, bq, Qbf, nullptr,
      C, C, 0, 0, 0, 1.f, 0);
  gemm_bf16_128x128<<<gProj, 256, 0, stream>>>(xbf, wkbf, bk, Kbf, nullptr,
      C, C, 0, 0, 0, 1.f, 0);
  gemm_bf16_128x128<<<gProj, 256, 0, stream>>>(xbf, wvbf, bv, Vt, nullptr,
      C, C, 0, 0, 0, 1.f, 1);

  // 3) scores S = QK^T / sqrt(C), causal  (per-batch strided)
  dim3 gScr(T / 128, T / 128, Bz);
  gemm_bf16_128x128<<<gScr, 256, 0, stream>>>(Qbf, Kbf, nullptr, nullptr, Sf,
      T, C, (long long)T * C, (long long)T * C, (long long)T * T,
      0.03125f /* 1/sqrt(1024) */, 2);

  // 4) softmax rows -> P (bf16)
  softmax_rows<<<Bz * T, 256, 0, stream>>>(Sf, Pbf);

  // 5) O = P @ V   (B-operand is Vt[b], K-major along s)
  dim3 gAV(C / 128, T / 128, Bz);
  gemm_bf16_128x128<<<gAV, 256, 0, stream>>>(Pbf, Vt, nullptr, Obf, nullptr,
      C, T, (long long)T * T, (long long)C * T, (long long)T * C, 1.f, 3);

  // 6) out = O @ Wf^T + bf  (f32 output)
  gemm_bf16_128x128<<<gProj, 256, 0, stream>>>(Obf, wfbf, bfp, nullptr,
      (float*)d_out, C, C, 0, 0, 0, 1.f, 4);
}
